// PEM_66692252172459
// MI455X (gfx1250) — matrix operand order repaired and checked by reading, not compiled, not run on hardware
//
#include <hip/hip_runtime.h>
#include <math.h>

#define TSC   100
#define NSAMP 32
#define NPOS  10000      // D*T
#define BATCH 4
#define KBIG  3200       // T*NSAMP
#define MBIG  2048       // B*512

typedef __attribute__((ext_vector_type(16))) __bf16 v16bf;
typedef __attribute__((ext_vector_type(8)))  __bf16 v8bf;
typedef __attribute__((ext_vector_type(8)))  float  v8f;

__device__ __forceinline__ unsigned short f2bf(float f) {
  union { float f; unsigned u; } v; v.f = f;
  unsigned r = (v.u + 0x7FFFu + ((v.u >> 16) & 1u)) >> 16;
  return (unsigned short)r;
}
__device__ __forceinline__ float bf2f(unsigned short h) {
  union { unsigned u; float f; } v; v.u = ((unsigned)h) << 16;
  return v.f;
}

// ---------------- stage 1: conv1d 256->128 k=3 pad=1 + ReLU ----------------
__global__ void k_conv1d(const float* __restrict__ x, const float* __restrict__ w,
                         const float* __restrict__ bias, float* __restrict__ h) {
  int idx = blockIdx.x * blockDim.x + threadIdx.x;
  if (idx >= BATCH * 128 * TSC) return;
  int t = idx % TSC, c = (idx / TSC) % 128, b = idx / (128 * TSC);
  float acc = bias[c];
  for (int ci = 0; ci < 256; ++ci) {
    const float* xp = x + ((size_t)b * 256 + ci) * TSC;
    const float* wp = w + ((size_t)c * 256 + ci) * 3;
    #pragma unroll
    for (int kk = 0; kk < 3; ++kk) {
      int tt = t + kk - 1;
      if (tt >= 0 && tt < TSC) acc += xp[tt] * wp[kk];
    }
  }
  h[idx] = acc > 0.f ? acc : 0.f;
}

// ---------------- stage 2: G[(b*512+o)][t*32+s] = sum_c w3d[o,c,s]*h[b,c,t] ----------------
__global__ void k_G(const float* __restrict__ w3d, const float* __restrict__ h,
                    unsigned short* __restrict__ G) {
  int idx = blockIdx.x * blockDim.x + threadIdx.x;
  if (idx >= MBIG * KBIG) return;
  int k = idx % KBIG, m = idx / KBIG;
  int s = k & 31, t = k >> 5;
  int o = m & 511, b = m >> 9;
  const float* wp = w3d + (size_t)o * 128 * 32 + s;
  const float* hp = h + (size_t)b * 128 * TSC + t;
  float acc = 0.f;
  for (int c = 0; c < 128; ++c) acc += wp[c * 32] * hp[c * TSC];
  G[idx] = f2bf(acc);
}

// ---------------- mask [t,s,d,u] -> bf16 column-major B [n=(d,u)][k=(t,s)] ----------------
// writes are fully coalesced (idx-linear); strided reads stay L2-resident (128MB <= 192MB L2)
__global__ void k_maskB(const float* __restrict__ mask, unsigned short* __restrict__ Bm) {
  long long idx = (long long)blockIdx.x * blockDim.x + threadIdx.x;
  if (idx >= (long long)KBIG * NPOS) return;
  int n = (int)(idx / KBIG), k = (int)(idx % KBIG);
  Bm[idx] = f2bf(mask[(size_t)k * NPOS + n]);
}

// ---------------- plain f32 -> bf16 copy (w2d) ----------------
__global__ void k_cvt(const float* __restrict__ in, unsigned short* __restrict__ out, int n) {
  int idx = blockIdx.x * blockDim.x + threadIdx.x;
  if (idx < n) out[idx] = f2bf(in[idx]);
}

// ---------------- wf [ko][c][dy][dx] -> per-tap row-major A [tap][ko][c] ----------------
__global__ void k_cvt_wf(const float* __restrict__ in, unsigned short* __restrict__ out) {
  int idx = blockIdx.x * blockDim.x + threadIdx.x;
  if (idx >= 9 * 128 * 128) return;
  int tap = idx / 16384, rem = idx % 16384;
  int ko = rem / 128, c = rem % 128;
  int dy = tap / 3, dx = tap % 3;
  out[idx] = f2bf(in[(((size_t)ko * 128 + c) * 3 + dy) * 3 + dx]);
}

__device__ __forceinline__ v16bf cat8(v8bf lo, v8bf hi) {
  return __builtin_shufflevector(lo, hi, 0,1,2,3,4,5,6,7,8,9,10,11,12,13,14,15);
}

// ---------------- BIG GEMM: C[2048,10000] = G[2048,3200] x maskB^T, +b3d, ReLU -> bf16 colmaj
// register-blocked 32x80 per wave: 2 A-frags x 5 B-frags -> 10 accumulators
__global__ void k_gemm_m3(const unsigned short* __restrict__ A, const unsigned short* __restrict__ Bm,
                          const float* __restrict__ b3d, unsigned short* __restrict__ m3bf) {
  int wave = (blockIdx.x * blockDim.x + threadIdx.x) >> 5;  // 8000 waves exactly
  int lane = threadIdx.x & 31;
  int nb = wave >> 6;          // 0..124 : 80-column block (consecutive waves share it -> L2 reuse)
  int mb = wave & 63;          // 0..63  : 32-row block
  int m0 = mb * 32, n0 = nb * 80;
  int hf = lane >> 4, r = lane & 15;
  const unsigned short* a0 = A + (size_t)(m0 + r) * KBIG + hf * 16;
  const unsigned short* a1 = a0 + (size_t)16 * KBIG;
  const unsigned short* bp0 = Bm + (size_t)(n0 + r) * KBIG + hf * 16;
  v8f acc[10] = {};
  for (int kb = 0; kb < KBIG; kb += 32) {
    v16bf af0 = cat8(*(const v8bf*)(a0 + kb), *(const v8bf*)(a0 + kb + 8));
    v16bf af1 = cat8(*(const v8bf*)(a1 + kb), *(const v8bf*)(a1 + kb + 8));
    #pragma unroll
    for (int j = 0; j < 5; ++j) {
      const unsigned short* bp = bp0 + (size_t)j * 16 * KBIG + kb;
      v16bf bfr = *(const v16bf*)bp;
      __builtin_prefetch(bp + 512, 0, 0);   // 1KB ahead in this B stream (global_prefetch_b8)
      acc[j]     = __builtin_amdgcn_wmma_f32_16x16x32_bf16(false, af0, false, bfr, (short)0, acc[j],     false, false);
      acc[5 + j] = __builtin_amdgcn_wmma_f32_16x16x32_bf16(false, af1, false, bfr, (short)0, acc[5 + j], false, false);
    }
  }
  int n_base = n0 + r;
  #pragma unroll
  for (int i = 0; i < 2; ++i) {
    #pragma unroll
    for (int j = 0; j < 5; ++j) {
      #pragma unroll
      for (int e = 0; e < 8; ++e) {
        int m = m0 + i * 16 + e + hf * 8;
        int o = m & 511, b = m >> 9;
        int n = n_base + j * 16;
        float v = acc[i * 5 + j][e] + b3d[o];
        v = v > 0.f ? v : 0.f;
        m3bf[((size_t)b * NPOS + n) * 512 + o] = f2bf(v);
      }
    }
  }
}

// ---------------- 1x1 conv GEMM: m2[b] = w2d[128,512] x m3[b], +b2d, ReLU -> bf16 colmaj
// full-M blocking: 8 A-frags share one B fragment -> 8 accumulators
__global__ void k_gemm_m2(const unsigned short* __restrict__ A, const unsigned short* __restrict__ Bm,
                          const float* __restrict__ b2d, unsigned short* __restrict__ m2bf) {
  int wave = (blockIdx.x * blockDim.x + threadIdx.x) >> 5;
  if (wave >= BATCH * 625) return;   // wave-uniform guard (no WMMA past this point)
  int lane = threadIdx.x & 31;
  int b = wave / 625, nt = wave % 625;
  int n0 = nt << 4;
  int hf = lane >> 4, r = lane & 15;
  const unsigned short* a0 = A + (size_t)r * 512 + hf * 16;
  const unsigned short* bp = Bm + (size_t)b * NPOS * 512 + (size_t)(n0 + r) * 512 + hf * 16;
  v8f acc[8] = {};
  for (int kb = 0; kb < 512; kb += 32) {
    v16bf bfr = *(const v16bf*)(bp + kb);
    #pragma unroll
    for (int i = 0; i < 8; ++i) {
      const unsigned short* ap = a0 + (size_t)i * 16 * 512 + kb;
      v16bf af = cat8(*(const v8bf*)ap, *(const v8bf*)(ap + 8));
      acc[i] = __builtin_amdgcn_wmma_f32_16x16x32_bf16(false, af, false, bfr, (short)0, acc[i], false, false);
    }
  }
  int n = n0 + r;
  #pragma unroll
  for (int i = 0; i < 8; ++i) {
    #pragma unroll
    for (int e = 0; e < 8; ++e) {
      int m = i * 16 + e + hf * 8;
      float v = acc[i][e] + b2d[m];
      v = v > 0.f ? v : 0.f;
      m2bf[((size_t)b * NPOS + n) * 128 + m] = f2bf(v);
    }
  }
}

// ---------------- 3x3 conv: 9 accumulating K=128 implicit GEMMs, full-M blocked ----------------
__global__ void k_conv3x3(const unsigned short* __restrict__ src, const unsigned short* __restrict__ wt,
                          const float* __restrict__ bias, unsigned short* __restrict__ dst) {
  int wave = (blockIdx.x * blockDim.x + threadIdx.x) >> 5;
  if (wave >= BATCH * 625) return;   // wave-uniform guard
  int lane = threadIdx.x & 31;
  int b = wave / 625, nt = wave % 625;
  int n0 = nt << 4;
  int hf = lane >> 4, r = lane & 15;
  int n = n0 + r;
  int d = n / TSC, u = n % TSC;
  const unsigned short* sb = src + (size_t)b * NPOS * 128;
  v8f acc[8] = {};
  for (int tap = 0; tap < 9; ++tap) {
    int dy = tap / 3 - 1, dx = tap % 3 - 1;
    int dd = d + dy, uu = u + dx;
    bool ok = ((unsigned)dd < (unsigned)TSC) && ((unsigned)uu < (unsigned)TSC);
    const unsigned short* colp = sb + (size_t)(dd * TSC + uu) * 128 + hf * 16;
    const unsigned short* wp0 = wt + (size_t)tap * 16384 + (size_t)r * 128 + hf * 16;
    #pragma unroll
    for (int kb = 0; kb < 128; kb += 32) {
      v16bf bfr = {};                      // zero fragment implements the conv padding
      if (ok) bfr = *(const v16bf*)(colp + kb);
      #pragma unroll
      for (int i = 0; i < 8; ++i) {
        const unsigned short* wp = wp0 + (size_t)i * 16 * 128 + kb;
        v16bf af = cat8(*(const v8bf*)wp, *(const v8bf*)(wp + 8));
        acc[i] = __builtin_amdgcn_wmma_f32_16x16x32_bf16(false, af, false, bfr, (short)0, acc[i], false, false);
      }
    }
  }
  #pragma unroll
  for (int i = 0; i < 8; ++i) {
    #pragma unroll
    for (int e = 0; e < 8; ++e) {
      int m = i * 16 + e + hf * 8;
      float v = acc[i][e] + bias[m];
      v = v > 0.f ? v : 0.f;
      dst[((size_t)b * NPOS + n) * 128 + m] = f2bf(v);
    }
  }
}

// ---------------- head: out[b,j,n] = sigmoid(wf3[j]·f[b,:,n] + bf3[j]) ----------------
__global__ void k_head(const unsigned short* __restrict__ f, const float* __restrict__ wf3,
                       const float* __restrict__ bf3, float* __restrict__ out) {
  int idx = blockIdx.x * blockDim.x + threadIdx.x;
  if (idx >= BATCH * NPOS) return;
  int b = idx / NPOS, n = idx % NPOS;
  const unsigned short* fp = f + ((size_t)b * NPOS + n) * 128;
  float s0 = bf3[0], s1 = bf3[1];
  for (int c = 0; c < 128; ++c) {
    float v = bf2f(fp[c]);
    s0 += v * wf3[c];
    s1 += v * wf3[128 + c];
  }
  out[((size_t)b * 2 + 0) * NPOS + n] = 1.f / (1.f + __expf(-s0));
  out[((size_t)b * 2 + 1) * NPOS + n] = 1.f / (1.f + __expf(-s1));
}

extern "C" void kernel_launch(void* const* d_in, const int* in_sizes, int n_in,
                              void* d_out, int out_size, void* d_ws, size_t ws_size,
                              hipStream_t stream) {
  const float* x    = (const float*)d_in[0];
  const float* w_rd = (const float*)d_in[1];
  const float* b_rd = (const float*)d_in[2];
  const float* w3d  = (const float*)d_in[3];
  const float* b3d  = (const float*)d_in[4];
  const float* w2d  = (const float*)d_in[5];
  const float* b2d  = (const float*)d_in[6];
  const float* wf1  = (const float*)d_in[7];
  const float* bf1  = (const float*)d_in[8];
  const float* wf2  = (const float*)d_in[9];
  const float* bf2  = (const float*)d_in[10];
  const float* wf3  = (const float*)d_in[11];
  const float* bf3  = (const float*)d_in[12];
  const float* mask = (const float*)d_in[13];
  float* out = (float*)d_out;

  char* ws = (char*)d_ws;
  // workspace carving (all offsets 256B aligned)
  float*          h      = (float*)(ws + 0);                      //   204,800 B
  unsigned short* Gbf    = (unsigned short*)(ws + 204800);        //  13,107,200 B
  unsigned short* maskB  = (unsigned short*)(ws + 13312000);      //  64,000,000 B
  unsigned short* m3bf   = (unsigned short*)(ws + 77312000);      //  40,960,000 B
  unsigned short* w2dbf  = (unsigned short*)(ws + 118272000);     //     131,072 B
  unsigned short* m2bf   = (unsigned short*)(ws + 118403072);     //  10,240,000 B
  unsigned short* wf1bf  = (unsigned short*)(ws + 128643072);     //     294,912 B
  unsigned short* wf2bf  = (unsigned short*)(ws + 128937984);     //     294,912 B
  unsigned short* f1bf   = (unsigned short*)(ws + 129232896);     //  10,240,000 B
  unsigned short* f2bf_b = (unsigned short*)(ws + 139472896);     //  10,240,000 B  (end ~149.7 MB)

  // stage 1: reduce_dim conv1d + ReLU
  k_conv1d<<<200, 256, 0, stream>>>(x, w_rd, b_rd, h);
  // stage 2: G = w3d x h  (A operand of the big GEMM, bf16 row-major)
  k_G<<<25600, 256, 0, stream>>>(w3d, h, Gbf);
  // mask -> bf16 column-major B operand
  k_maskB<<<125000, 256, 0, stream>>>(mask, maskB);
  // weight conversions
  k_cvt<<<256, 256, 0, stream>>>(w2d, w2dbf, 128 * 512);
  k_cvt_wf<<<576, 256, 0, stream>>>(wf1, wf1bf);
  k_cvt_wf<<<576, 256, 0, stream>>>(wf2, wf2bf);
  // big fused sampling+conv3d GEMM (131 GFLOP), 32x80 per wave -> m3 (bf16 col-major per batch)
  k_gemm_m3<<<1000, 256, 0, stream>>>(Gbf, maskB, b3d, m3bf);
  // 1x1 conv GEMM 512->128, 128x16 per wave
  k_gemm_m2<<<313, 256, 0, stream>>>(w2dbf, m3bf, b2d, m2bf);
  // two 3x3 convs as implicit WMMA GEMMs, 128x16 per wave
  k_conv3x3<<<313, 256, 0, stream>>>(m2bf, wf1bf, bf1, f1bf);
  k_conv3x3<<<313, 256, 0, stream>>>(f1bf, wf2bf, bf2, f2bf_b);
  // sigmoid head
  k_head<<<157, 256, 0, stream>>>(f2bf_b, wf3, bf3, out);

  (void)in_sizes; (void)n_in; (void)out_size; (void)ws_size;
}
